// URLTextProcessor_48206712930853
// MI455X (gfx1250) — matrix operand order repaired and checked by reading, not compiled, not run on hardware
//
#include <hip/hip_runtime.h>
#include <hip/hip_bf16.h>

// Problem constants (from reference)
#define VOCAB 100
#define EMB   32
#define HID   64
#define NOUT  32
#define BSZ   8192
#define TT    200

typedef __attribute__((ext_vector_type(16))) __bf16 v16bf;
typedef __attribute__((ext_vector_type(8)))  float  v8f;
typedef __attribute__((ext_vector_type(4)))  float  f4;

#define WAVES    4
#define BLOCK    (WAVES * 32)
#define MROWS    16                  // batch rows per wave (one WMMA M tile)
#define MPERBLK  (WAVES * MROWS)     // 64 batch rows per workgroup
#define HSTRIDE  72                  // padded LDS stride (halves) for h staging

// ---- LDS layout (bytes), all 16B aligned ----
constexpr int OFF_WIH0 = 0;                                  // 256x32 bf16 = 16384
constexpr int OFF_WHH0 = OFF_WIH0 + 256 * 32 * 2;            // 256x64 bf16 = 32768
constexpr int OFF_WIH1 = OFF_WHH0 + 256 * 64 * 2;
constexpr int OFF_WHH1 = OFF_WIH1 + 256 * 64 * 2;
constexpr int OFF_WFC  = OFF_WHH1 + 256 * 64 * 2;            // 32x64 bf16 = 4096
constexpr int OFF_EMB  = OFF_WFC  + 32 * 64 * 2;             // 100x32 bf16 = 6400
constexpr int OFF_B0   = OFF_EMB  + VOCAB * EMB * 2;         // 256 f32
constexpr int OFF_B1   = OFF_B0   + 256 * 4;                 // 256 f32
constexpr int OFF_BFC  = OFF_B1   + 256 * 4;                 // 32 f32
constexpr int OFF_STG  = OFF_BFC  + 32 * 4;                  // 8 staging tiles
constexpr int SMEM_BYTES = OFF_STG + WAVES * 2 * 16 * HSTRIDE * 2;  // ~142.4 KB

// ---------------- fragment helpers ----------------

__device__ __forceinline__ v16bf zero16() {
  v16bf r;
  __builtin_memset(&r, 0, sizeof(r));
  return r;
}

// B-matrix fragment (16-bit, 32x16 tile of a KxN matrix stored [N][K] row-major).
// CDNA5 dense 16-bit B layout: lanes 0-15 hold K=0..7,16..23; lanes 16-31 K=8..15,24..31.
__device__ __forceinline__ v16bf ld_bfrag(const __bf16* W, int K, int ntile, int kofs, int lane) {
  int n  = ntile * 16 + (lane & 15);
  int kb = kofs + ((lane >> 4) << 3);
  const __bf16* p = W + n * K + kb;
  v16bf r;
  ((f4*)&r)[0] = *(const f4*)(p);
  ((f4*)&r)[1] = *(const f4*)(p + 16);
  return r;
}

// A-matrix fragment (16-bit, 16x32 tile, row-major source with given stride in halves).
// CDNA5 layout: lane row = lane&15; kb = (lane>=16)?8:0 (+kofs);
// halves[0..7] = A[row][kb..kb+7], halves[8..15] = A[row][kb+16..kb+23].
__device__ __forceinline__ v16bf ld_afrag(const __bf16* A, int stride, int kofs, int lane) {
  int row = lane & 15;
  int kb  = kofs + ((lane >> 4) << 3);
  const __bf16* p = A + row * stride + kb;
  v16bf r;
  ((f4*)&r)[0] = *(const f4*)(p);
  ((f4*)&r)[1] = *(const f4*)(p + 16);
  return r;
}

// Store a C/D-layout (f32) 16x16 tile as bf16 into LDS row-major buffer.
// C layout: VGPR r -> M = r + 8*(lane>>4), N = lane&15.
__device__ __forceinline__ void st_ctile_bf16(__bf16* buf, int stride, int ncol0,
                                              const v8f v, int lane) {
  int nc = ncol0 + (lane & 15);
  int mb = (lane >> 4) * 8;
#pragma unroll
  for (int r = 0; r < 8; ++r) buf[(mb + r) * stride + nc] = (__bf16)v[r];
}

// ---------------- activations (overflow-graceful) ----------------

__device__ __forceinline__ float sig1(float x) { return 1.0f / (1.0f + __expf(-x)); }
__device__ __forceinline__ float tanh1(float x) {
  float ax = __builtin_fabsf(x);
  float e  = __expf(2.0f * ax);           // inf for big ax -> 2/inf = 0 -> +/-1
  float t  = 1.0f - 2.0f / (e + 1.0f);
  return __builtin_copysignf(t, x);
}
__device__ __forceinline__ v8f sig8(v8f x) {
  v8f r;
#pragma unroll
  for (int i = 0; i < 8; ++i) r[i] = sig1(x[i]);
  return r;
}
__device__ __forceinline__ v8f tanh8(v8f x) {
  v8f r;
#pragma unroll
  for (int i = 0; i < 8; ++i) r[i] = tanh1(x[i]);
  return r;
}

__device__ __forceinline__ v8f splat8(float v) {
  v8f r = {v, v, v, v, v, v, v, v};
  return r;
}

#define WMMA_BF16(Afrag, Bfrag, Cacc) \
  __builtin_amdgcn_wmma_f32_16x16x32_bf16(false, (Afrag), false, (Bfrag), (short)0, (Cacc), false, false)

// ---------------- kernel ----------------

__global__ void __launch_bounds__(BLOCK, 1)
lstm2_wmma_kernel(const int* __restrict__ x, const float* __restrict__ emb,
                  const float* __restrict__ Wih0, const float* __restrict__ Whh0,
                  const float* __restrict__ bih0, const float* __restrict__ bhh0,
                  const float* __restrict__ Wih1, const float* __restrict__ Whh1,
                  const float* __restrict__ bih1, const float* __restrict__ bhh1,
                  const float* __restrict__ Wfc,  const float* __restrict__ bfc,
                  float* __restrict__ out) {
  extern __shared__ char smem[];
  __bf16* sWih0 = (__bf16*)(smem + OFF_WIH0);
  __bf16* sWhh0 = (__bf16*)(smem + OFF_WHH0);
  __bf16* sWih1 = (__bf16*)(smem + OFF_WIH1);
  __bf16* sWhh1 = (__bf16*)(smem + OFF_WHH1);
  __bf16* sWfc  = (__bf16*)(smem + OFF_WFC);
  __bf16* sEmb  = (__bf16*)(smem + OFF_EMB);
  float*  sB0   = (float*)(smem + OFF_B0);
  float*  sB1   = (float*)(smem + OFF_B1);
  float*  sBfc  = (float*)(smem + OFF_BFC);

  const int tid  = threadIdx.x;
  const int lane = tid & 31;
  const int wave = tid >> 5;

  // ---- one-time cooperative: convert weights/emb to bf16 in LDS, fuse biases ----
  for (int i = tid; i < 256 * 32; i += BLOCK) sWih0[i] = (__bf16)Wih0[i];
  for (int i = tid; i < 256 * 64; i += BLOCK) sWhh0[i] = (__bf16)Whh0[i];
  for (int i = tid; i < 256 * 64; i += BLOCK) sWih1[i] = (__bf16)Wih1[i];
  for (int i = tid; i < 256 * 64; i += BLOCK) sWhh1[i] = (__bf16)Whh1[i];
  for (int i = tid; i < 32 * 64;  i += BLOCK) sWfc[i]  = (__bf16)Wfc[i];
  for (int i = tid; i < VOCAB * EMB; i += BLOCK) sEmb[i] = (__bf16)emb[i];
  for (int i = tid; i < 256; i += BLOCK) {
    sB0[i] = bih0[i] + bhh0[i];
    sB1[i] = bih1[i] + bhh1[i];
  }
  for (int i = tid; i < 32; i += BLOCK) sBfc[i] = bfc[i];
  __syncthreads();

  // ---- loop-invariant register residency ----
  // Wih0 B-fragments (K=EMB=32 -> one fragment per N tile): 16 x 8 VGPRs.
  v16bf bx0[16];
#pragma unroll
  for (int nt = 0; nt < 16; ++nt) bx0[nt] = ld_bfrag(sWih0, EMB, nt, 0, lane);
  // Bias splats held as scalars in VGPRs (per-lane column value).
  float b0r[16], b1r[16];
#pragma unroll
  for (int nt = 0; nt < 16; ++nt) {
    b0r[nt] = sB0[nt * 16 + (lane & 15)];
    b1r[nt] = sB1[nt * 16 + (lane & 15)];
  }
  float bfcr[2];
#pragma unroll
  for (int nt = 0; nt < 2; ++nt) bfcr[nt] = sBfc[nt * 16 + (lane & 15)];

  // Per-wave private staging tiles (same-wave LDS store->load is in-order; no barrier needed)
  __bf16* h0buf = (__bf16*)(smem + OFF_STG) + wave * 2 * 16 * HSTRIDE;
  __bf16* h1buf = h0buf + 16 * HSTRIDE;

  const int mbase = blockIdx.x * MPERBLK + wave * MROWS;  // batch row base of this wave
  const int row   = lane & 15;
  const int* xrow = x + (mbase + row) * TT;

  // Recurrent state in registers
  v8f c0[4], c1[4];
#pragma unroll
  for (int j = 0; j < 4; ++j) { c0[j] = splat8(0.f); c1[j] = splat8(0.f); }
  v16bf ah0_lo = zero16(), ah0_hi = zero16();
  v16bf ah1_lo = zero16(), ah1_hi = zero16();

  // Software-prefetched token index (hides the global-load latency behind a
  // full timestep of WMMA work instead of stalling the serial recurrence).
  int idx = xrow[0];

#pragma unroll 1
  for (int t = 0; t < TT; ++t) {
    const int tn = (t + 1 < TT) ? (t + 1) : (TT - 1);
    const int idx_next = xrow[tn];

    // Embedding A-fragment (16x32 bf16) gathered from LDS emb table by token index
    const __bf16* ep = sEmb + idx * EMB + ((lane >> 4) << 3);
    v16bf ax;
    ((f4*)&ax)[0] = *(const f4*)(ep);
    ((f4*)&ax)[1] = *(const f4*)(ep + 16);

    // -------- layer 0: gates = x*Wih0^T + h0*Whh0^T + b0 --------
#pragma unroll
    for (int jb = 0; jb < 4; ++jb) {
      // Batch all Whh0 B-fragment loads for this jb (8 frags = 16 ds_load_b128),
      // then issue the 12 WMMAs densely behind a single dscnt wait.
      v16bf bl[4], bh[4];
#pragma unroll
      for (int g = 0; g < 4; ++g) {
        int nt = g * 4 + jb;
        bl[g] = ld_bfrag(sWhh0, HID, nt, 0,  lane);
        bh[g] = ld_bfrag(sWhh0, HID, nt, 32, lane);
      }
      v8f acc[4];
#pragma unroll
      for (int g = 0; g < 4; ++g) {               // PyTorch gate order i,f,g,o
        int nt = g * 4 + jb;
        v8f a = splat8(b0r[nt]);
        a = WMMA_BF16(ax,     bx0[nt], a);
        a = WMMA_BF16(ah0_lo, bl[g],   a);
        a = WMMA_BF16(ah0_hi, bh[g],   a);
        acc[g] = a;
      }
      v8f it = sig8(acc[0]);
      v8f ft = sig8(acc[1]);
      v8f gt = tanh8(acc[2]);
      v8f ot = sig8(acc[3]);
      c0[jb] = ft * c0[jb] + it * gt;
      v8f h  = ot * tanh8(c0[jb]);
      st_ctile_bf16(h0buf, HSTRIDE, jb * 16, h, lane);
    }
    ah0_lo = ld_afrag(h0buf, HSTRIDE, 0,  lane);
    ah0_hi = ld_afrag(h0buf, HSTRIDE, 32, lane);

    // -------- layer 1: gates = h0*Wih1^T + h1*Whh1^T + b1 --------
#pragma unroll
    for (int jb = 0; jb < 4; ++jb) {
      // Batch all 16 B-fragments for this jb, then 16 dense WMMAs.
      v16bf wi_lo[4], wi_hi[4], wh_lo[4], wh_hi[4];
#pragma unroll
      for (int g = 0; g < 4; ++g) {
        int nt = g * 4 + jb;
        wi_lo[g] = ld_bfrag(sWih1, HID, nt, 0,  lane);
        wi_hi[g] = ld_bfrag(sWih1, HID, nt, 32, lane);
        wh_lo[g] = ld_bfrag(sWhh1, HID, nt, 0,  lane);
        wh_hi[g] = ld_bfrag(sWhh1, HID, nt, 32, lane);
      }
      v8f acc[4];
#pragma unroll
      for (int g = 0; g < 4; ++g) {
        int nt = g * 4 + jb;
        v8f a = splat8(b1r[nt]);
        a = WMMA_BF16(ah0_lo, wi_lo[g], a);
        a = WMMA_BF16(ah0_hi, wi_hi[g], a);
        a = WMMA_BF16(ah1_lo, wh_lo[g], a);
        a = WMMA_BF16(ah1_hi, wh_hi[g], a);
        acc[g] = a;
      }
      v8f it = sig8(acc[0]);
      v8f ft = sig8(acc[1]);
      v8f gt = tanh8(acc[2]);
      v8f ot = sig8(acc[3]);
      c1[jb] = ft * c1[jb] + it * gt;
      v8f h  = ot * tanh8(c1[jb]);
      st_ctile_bf16(h1buf, HSTRIDE, jb * 16, h, lane);
    }
    ah1_lo = ld_afrag(h1buf, HSTRIDE, 0,  lane);
    ah1_hi = ld_afrag(h1buf, HSTRIDE, 32, lane);

    idx = idx_next;
  }

  // -------- FC head: relu(h1_last @ Wfc^T + bfc), OUT=32 -> 2 N tiles --------
#pragma unroll
  for (int nt = 0; nt < 2; ++nt) {
    v8f a = splat8(bfcr[nt]);
    a = WMMA_BF16(ah1_lo, ld_bfrag(sWfc, HID, nt, 0,  lane), a);
    a = WMMA_BF16(ah1_hi, ld_bfrag(sWfc, HID, nt, 32, lane), a);
    int nc = nt * 16 + (lane & 15);
    int mb = (lane >> 4) * 8;
#pragma unroll
    for (int r = 0; r < 8; ++r) {
      float v = a[r] > 0.f ? a[r] : 0.f;
      out[(mbase + mb + r) * NOUT + nc] = v;
    }
  }
}

extern "C" void kernel_launch(void* const* d_in, const int* in_sizes, int n_in,
                              void* d_out, int out_size, void* d_ws, size_t ws_size,
                              hipStream_t stream) {
  (void)in_sizes; (void)n_in; (void)out_size; (void)d_ws; (void)ws_size;
  hipFuncSetAttribute(reinterpret_cast<const void*>(lstm2_wmma_kernel),
                      hipFuncAttributeMaxDynamicSharedMemorySize, SMEM_BYTES);
  dim3 grid(BSZ / MPERBLK);   // 128 workgroups
  dim3 block(BLOCK);          // 128 threads = 4 waves
  lstm2_wmma_kernel<<<grid, block, SMEM_BYTES, stream>>>(
      (const int*)d_in[0],  (const float*)d_in[1],
      (const float*)d_in[2], (const float*)d_in[3],
      (const float*)d_in[4], (const float*)d_in[5],
      (const float*)d_in[6], (const float*)d_in[7],
      (const float*)d_in[8], (const float*)d_in[9],
      (const float*)d_in[10], (const float*)d_in[11],
      (float*)d_out);
}
